// FuzzySigmoidLayer_31456340476194
// MI455X (gfx1250) — matrix operand order repaired and checked by reading, not compiled, not run on hardware
//
#include <hip/hip_runtime.h>
#include <hip/hip_bf16.h>

// Problem constants (match reference setup_inputs()).
#define B_     16384
#define D_IN_  2048
#define D_OUT_ 2048
#define TOPK_  256
#define BN_EPS_ 1e-5f

// GEMM tiling
#define TILE_M 128
#define TILE_N 128
#define TILE_K 32
#define KSTEPS (D_IN_ / TILE_K)
#define LDSS   40   // LDS row stride in halfs: 32 data + 8 pad (80B, bank-conflict-free)

typedef __attribute__((ext_vector_type(16))) __bf16 v16bf;
typedef __attribute__((ext_vector_type(8)))  __bf16 v8bf;
typedef __attribute__((ext_vector_type(8)))  float  v8f;
typedef __attribute__((ext_vector_type(4)))  unsigned int v4u32;
typedef __attribute__((ext_vector_type(8)))  int v8i32;
typedef __attribute__((ext_vector_type(4)))  int v4i32;

#if defined(__has_builtin)
#if __has_builtin(__builtin_amdgcn_tensor_load_to_lds) && \
    __has_builtin(__builtin_amdgcn_s_wait_tensorcnt)
#define HAVE_TDM 1
#endif
#endif
#ifndef HAVE_TDM
#define HAVE_TDM 0
#endif

// ---------------------------------------------------------------------------
// Kernel 0: convert base_weight f32 -> bf16 (8 MiB bf16: L2-resident)
// ---------------------------------------------------------------------------
__global__ void k_wconv(const float* __restrict__ Wf, __hip_bfloat16* __restrict__ Wbf) {
  const size_t n = (size_t)D_OUT_ * D_IN_;
  const size_t stride = (size_t)gridDim.x * blockDim.x;
  for (size_t i = (size_t)blockIdx.x * blockDim.x + threadIdx.x; i < n; i += stride)
    Wbf[i] = __float2bfloat16(Wf[i]);
}

// ---------------------------------------------------------------------------
// Kernel 1: one 256-thread block per row.
//   mu = sigmoid(sharp*(x-center)); bitonic-sort (desc) 2048 (mu|col) keys in
//   LDS; ranks 0..255 weighted by linspace(1,0.1,256)/140.8 and scattered as
//   A[row,col] = bf16(w_r * mu_r * x[row,col]); rest of A row zeroed.
// ---------------------------------------------------------------------------
__global__ void __launch_bounds__(256) k_mu_owa(
    const float* __restrict__ x, const float* __restrict__ center,
    const float* __restrict__ sharp, float* __restrict__ mu_out,
    __hip_bfloat16* __restrict__ Abf) {
  __shared__ unsigned long long keys[D_IN_];
  __shared__ float xrow[D_IN_];
  const int row = blockIdx.x;
  const int tid = threadIdx.x;
  const float* __restrict__ xr = x + (size_t)row * D_IN_;
  __hip_bfloat16* __restrict__ Ar = Abf + (size_t)row * D_IN_;
  float* __restrict__ mur = mu_out + (size_t)row * D_IN_;

  for (int i = tid; i < D_IN_; i += 256) {
    float xv = xr[i];
    xrow[i] = xv;
    float t = sharp[i] * (xv - center[i]);
    float m = 1.0f / (1.0f + __expf(-t));          // in (0,1): uint-monotonic
    mur[i] = m;
    keys[i] = ((unsigned long long)__float_as_uint(m) << 32) | (unsigned)i;
    Ar[i] = __float2bfloat16(0.0f);                // zero the sparse row
  }
  __syncthreads();

  for (int k = 2; k <= D_IN_; k <<= 1) {           // bitonic, descending
    for (int j = k >> 1; j > 0; j >>= 1) {
      for (int i = tid; i < D_IN_; i += 256) {
        int ixj = i ^ j;
        if (ixj > i) {
          unsigned long long a = keys[i], b = keys[ixj];
          if ((a < b) == ((i & k) == 0)) { keys[i] = b; keys[ixj] = a; }
        }
      }
      __syncthreads();
    }
  }

  if (tid < TOPK_) {
    unsigned long long key = keys[tid];
    int   col = (int)(key & 0xffffffffull);
    float m   = __uint_as_float((unsigned)(key >> 32));
    float w   = (1.0f - 0.9f * (float)tid * (1.0f / 255.0f)) * (1.0f / 140.8f);
    Ar[col] = __float2bfloat16(w * m * xrow[col]);
  }
}

// ---------------------------------------------------------------------------
// TDM: build D# and issue TENSOR_LOAD_TO_LDS for a [128-row x 32-half] 2D tile
// of a row-major [*, 2048] bf16 matrix, padded 4 DWORDs every 16 DWORDs on the
// LDS side (=> LDSS stride). Tracked with TENSORcnt.
// ---------------------------------------------------------------------------
#if HAVE_TDM
__device__ __forceinline__ void tdm_tile_load(const __bf16* gsrc, unsigned lds_off) {
  unsigned long long ga = (unsigned long long)(uintptr_t)gsrc;
  v4u32 g0;
  g0[0] = 1u;                                       // count=1, user descriptor
  g0[1] = lds_off;                                  // lds_addr (bytes)
  g0[2] = (unsigned)ga;                             // global_addr[31:0]
  g0[3] = (unsigned)((ga >> 32) & 0x01FFFFFFu) | (2u << 30);  // addr[56:32]|type=2
  v8i32 g1;
  g1[0] = (int)((1u << 16)        // data_size = 2 bytes
              | (1u << 20)        // pad_enable
              | (3u << 22)        // pad_interval: 16 DWORDs
              | (3u << 25));      // pad_amount: 4 DWORDs
  g1[1] = (int)(2048u << 16);     // tensor_dim0 = 2048 (bits 79:48, low half)
  g1[2] = (int)(2048u << 16);     // dim0 hi=0 | tensor_dim1 = 2048 (low half)
  g1[3] = (int)((unsigned)TILE_K << 16);  // tensor_dim1 hi=0 | tile_dim0 = 32
  g1[4] = TILE_M;                 // tile_dim1 = 128 rows, tile_dim2 = 0
  g1[5] = 2048;                   // tensor_dim0_stride = 2048 elements
  g1[6] = 0;                      // stride0 hi | tensor_dim1_stride lo
  g1[7] = 0;
  v4i32 gz = {0, 0, 0, 0};
#if __clang_major__ >= 23
  v8i32 gz8 = {0, 0, 0, 0, 0, 0, 0, 0};
  __builtin_amdgcn_tensor_load_to_lds(g0, g1, gz, gz, gz8, 0);
#else
  __builtin_amdgcn_tensor_load_to_lds(g0, g1, gz, gz, 0);
#endif
}
#endif

// Load a 16-half WMMA fragment from LDS as two 16B chunks (+0, +step halfs).
__device__ __forceinline__ v16bf ld_frag(const __bf16* p, int step) {
  union { struct { v8bf lo; v8bf hi; } p2; v16bf v; } u;
  u.p2.lo = *(const v8bf*)(p);
  u.p2.hi = *(const v8bf*)(p + step);
  return u.v;
}

// One 32-deep k-step for a 32x64 wave tile: batch all 12 ds_load_b128s
// (clause-friendly), then 8 back-to-back in-place WMMAs.
__device__ __forceinline__ void compute_step(
    const __bf16* __restrict__ Abase, const __bf16* __restrict__ Wbase,
    int wr, int wc, int l15, int laneHi, v8f acc[2][4]) {
  // A frag (16x32): lanes0-15 K0..15, lanes16-31 K16..31 (contig 16)
  v16bf a0 = ld_frag(Abase + (wr + l15) * LDSS + (laneHi << 4), 8);
  v16bf a1 = ld_frag(Abase + (wr + 16 + l15) * LDSS + (laneHi << 4), 8);
  // B frags (32x16): lane=N; lanes0-15 K0..15, lanes16-31 K16..31 (contig 16)
  v16bf b0 = ld_frag(Wbase + (wc + 0  + l15) * LDSS + (laneHi << 4), 8);
  v16bf b1 = ld_frag(Wbase + (wc + 16 + l15) * LDSS + (laneHi << 4), 8);
  v16bf b2 = ld_frag(Wbase + (wc + 32 + l15) * LDSS + (laneHi << 4), 8);
  v16bf b3 = ld_frag(Wbase + (wc + 48 + l15) * LDSS + (laneHi << 4), 8);
  acc[0][0] = __builtin_amdgcn_wmma_f32_16x16x32_bf16(false, a0, false, b0, (short)0, acc[0][0], false, false);
  acc[1][0] = __builtin_amdgcn_wmma_f32_16x16x32_bf16(false, a1, false, b0, (short)0, acc[1][0], false, false);
  acc[0][1] = __builtin_amdgcn_wmma_f32_16x16x32_bf16(false, a0, false, b1, (short)0, acc[0][1], false, false);
  acc[1][1] = __builtin_amdgcn_wmma_f32_16x16x32_bf16(false, a1, false, b1, (short)0, acc[1][1], false, false);
  acc[0][2] = __builtin_amdgcn_wmma_f32_16x16x32_bf16(false, a0, false, b2, (short)0, acc[0][2], false, false);
  acc[1][2] = __builtin_amdgcn_wmma_f32_16x16x32_bf16(false, a1, false, b2, (short)0, acc[1][2], false, false);
  acc[0][3] = __builtin_amdgcn_wmma_f32_16x16x32_bf16(false, a0, false, b3, (short)0, acc[0][3], false, false);
  acc[1][3] = __builtin_amdgcn_wmma_f32_16x16x32_bf16(false, a1, false, b3, (short)0, acc[1][3], false, false);
}

// ---------------------------------------------------------------------------
// Kernel 2: WMMA GEMM y = relu(A @ W^T + bias).
// 256 thr = 8 waves (4x2); block tile 128x128, wave tile 32x64.
// TDM double-buffers A/W k-slices into padded LDS; wave 0 drives the DMA
// (uniform scalar branch), s_wait_tensorcnt + barrier publishes each buffer.
// ---------------------------------------------------------------------------
__global__ void __launch_bounds__(256) k_gemm(
    const __bf16* __restrict__ A, const __bf16* __restrict__ W,
    const float* __restrict__ bias, float* __restrict__ y) {
  __shared__ __align__(16) __bf16 As[2][TILE_M * LDSS];
  __shared__ __align__(16) __bf16 Ws[2][TILE_N * LDSS];

  const int tid    = threadIdx.x;
  const int lane   = tid & 31;
  const int wave   = tid >> 5;
  const int l15    = lane & 15;
  const int laneHi = lane >> 4;
  const int wr     = (wave & 3) << 5;     // wave row offset in tile (32 rows)
  const int wc     = (wave >> 2) << 6;    // wave col offset in tile (64 cols)
  const int rowBase = blockIdx.y * TILE_M;
  const int colBase = blockIdx.x * TILE_N;

  const __bf16* __restrict__ Ag = A + (size_t)rowBase * D_IN_;
  const __bf16* __restrict__ Wg = W + (size_t)colBase * D_IN_;

  v8f acc[2][4] = {};

#if HAVE_TDM
  // Uniform (SGPR) branch condition: avoids saveexec regions in the k-loop.
  const int isWave0 = (__builtin_amdgcn_readfirstlane(wave) == 0);

  const unsigned offA0 = (unsigned)(uintptr_t)(void*)&As[0][0];
  const unsigned offA1 = (unsigned)(uintptr_t)(void*)&As[1][0];
  const unsigned offW0 = (unsigned)(uintptr_t)(void*)&Ws[0][0];
  const unsigned offW1 = (unsigned)(uintptr_t)(void*)&Ws[1][0];

  if (isWave0) {                          // wave 0 drives the TDM
    tdm_tile_load(Ag, offA0);
    tdm_tile_load(Wg, offW0);
  }
  for (int i = 0; i < KSTEPS; ++i) {
    const int buf = i & 1;
    if (isWave0) {
      if (i + 1 < KSTEPS) {
        const int kn = (i + 1) * TILE_K;
        tdm_tile_load(Ag + kn, buf ? offA0 : offA1);
        tdm_tile_load(Wg + kn, buf ? offW0 : offW1);
        __builtin_amdgcn_s_wait_tensorcnt(2);  // current pair done; next in flight
      } else {
        __builtin_amdgcn_s_wait_tensorcnt(0);
      }
    }
    __syncthreads();                      // publish buf to all 8 waves
    compute_step(&As[buf][0], &Ws[buf][0], wr, wc, l15, laneHi, acc);
    __syncthreads();                      // buf free for overwrite next iter
  }
#else
  // Fallback: register staging global -> LDS (single buffer).
  for (int i = 0; i < KSTEPS; ++i) {
    const int k0 = i * TILE_K;
    for (int u = tid; u < TILE_M * 4; u += 256) {
      const int r = u >> 2, c = (u & 3) << 3;
      *(v8bf*)(&As[0][r * LDSS + c]) = *(const v8bf*)(Ag + (size_t)r * D_IN_ + k0 + c);
      *(v8bf*)(&Ws[0][r * LDSS + c]) = *(const v8bf*)(Wg + (size_t)r * D_IN_ + k0 + c);
    }
    __syncthreads();
    compute_step(&As[0][0], &Ws[0][0], wr, wc, l15, laneHi, acc);
    __syncthreads();
  }
#endif

  // C/D layout: VGPR r -> M = r (lanes 0-15) | 8+r (lanes 16-31); N = lane%16.
  const int rOff = laneHi << 3;
#pragma unroll
  for (int st = 0; st < 2; ++st) {
#pragma unroll
    for (int nt = 0; nt < 4; ++nt) {
      const int col = colBase + wc + (nt << 4) + l15;
      const float bc = bias[col];
#pragma unroll
      for (int r = 0; r < 8; ++r) {
        float v = acc[st][nt][r] + bc;
        v = v > 0.0f ? v : 0.0f;          // relu
        y[(size_t)(rowBase + wr + (st << 4) + rOff + r) * D_OUT_ + col] = v;
      }
    }
  }
}

// ---------------------------------------------------------------------------
// Deterministic two-stage per-column BN stats; gamma/beta folded to scale/shift;
// in-place normalize of y.
// ---------------------------------------------------------------------------
__global__ void k_stats1(const float* __restrict__ y,
                         float* __restrict__ psum, float* __restrict__ psq) {
  const int col = blockIdx.x * 256 + threadIdx.x;
  const int r0  = blockIdx.y * (B_ / 32);
  float s = 0.0f, q = 0.0f;
  for (int r = 0; r < B_ / 32; ++r) {
    float v = y[(size_t)(r0 + r) * D_OUT_ + col];
    s += v; q += v * v;
  }
  psum[(size_t)blockIdx.y * D_OUT_ + col] = s;
  psq [(size_t)blockIdx.y * D_OUT_ + col] = q;
}

__global__ void k_stats2(const float* __restrict__ psum, const float* __restrict__ psq,
                         const float* __restrict__ gamma, const float* __restrict__ beta,
                         float* __restrict__ scale, float* __restrict__ shift) {
  const int col = blockIdx.x * 256 + threadIdx.x;
  float s = 0.0f, q = 0.0f;
  for (int p = 0; p < 32; ++p) {
    s += psum[(size_t)p * D_OUT_ + col];
    q += psq [(size_t)p * D_OUT_ + col];
  }
  const float mean = s * (1.0f / (float)B_);
  const float var  = q * (1.0f / (float)B_) - mean * mean;
  const float sc   = gamma[col] * rsqrtf(var + BN_EPS_);
  scale[col] = sc;
  shift[col] = beta[col] - mean * sc;
}

__global__ void k_norm(float* __restrict__ y, const float* __restrict__ scale,
                       const float* __restrict__ shift) {
  const size_t n = (size_t)B_ * D_OUT_;
  const size_t stride = (size_t)gridDim.x * blockDim.x;
  for (size_t i = (size_t)blockIdx.x * blockDim.x + threadIdx.x; i < n; i += stride) {
    const int col = (int)(i & (D_OUT_ - 1));
    y[i] = y[i] * scale[col] + shift[col];
  }
}

// ---------------------------------------------------------------------------
// Launch. Workspace: [0,8MiB) W bf16 | [8,72MiB) A bf16 | then psum/psq (32x2048
// f32 each) | scale | shift.  d_out: y [B*D_OUT] f32, then mu [B*D_IN] f32.
// ---------------------------------------------------------------------------
extern "C" void kernel_launch(void* const* d_in, const int* in_sizes, int n_in,
                              void* d_out, int out_size, void* d_ws, size_t ws_size,
                              hipStream_t stream) {
  const float* x      = (const float*)d_in[0];
  const float* Wf     = (const float*)d_in[1];
  const float* bias   = (const float*)d_in[2];
  const float* center = (const float*)d_in[3];
  const float* sharp  = (const float*)d_in[4];
  const float* gamma  = (const float*)d_in[5];
  const float* beta   = (const float*)d_in[6];
  // d_in[7] = top_k (== 256, baked into TOPK_)

  float* y  = (float*)d_out;
  float* mu = y + (size_t)B_ * D_OUT_;

  char* ws = (char*)d_ws;
  __hip_bfloat16* Wbf = (__hip_bfloat16*)ws;
  __hip_bfloat16* Abf = (__hip_bfloat16*)(ws + (size_t)8 * 1024 * 1024);
  float* psum  = (float*)(ws + (size_t)72 * 1024 * 1024);
  float* psq   = psum + 32 * D_OUT_;
  float* scale = psq  + 32 * D_OUT_;
  float* shift = scale + D_OUT_;

  k_wconv<<<4096, 256, 0, stream>>>(Wf, Wbf);
  k_mu_owa<<<B_, 256, 0, stream>>>(x, center, sharp, mu, Abf);
  k_gemm<<<dim3(D_OUT_ / TILE_N, B_ / TILE_M), 256, 0, stream>>>(
      (const __bf16*)Abf, (const __bf16*)Wbf, bias, y);
  k_stats1<<<dim3(D_OUT_ / 256, 32), 256, 0, stream>>>(y, psum, psq);
  k_stats2<<<D_OUT_ / 256, 256, 0, stream>>>(psum, psq, gamma, beta, scale, shift);
  k_norm<<<4096, 256, 0, stream>>>(y, scale, shift);
}
